// RNN_GRU_45251775430852
// MI455X (gfx1250) — matrix operand order repaired and checked by reading, not compiled, not run on hardware
//
#include <hip/hip_runtime.h>

// GRU fused kernel for gfx1250 (MI455X).
// B=4096, T=120, I=64, H=128, gates = 3H = 384 (r, z, n order).
//
// 256 blocks x 16 batch rows; each block keeps W_ih/W_hh (bf16, 144 KB) LDS-
// resident and runs all T=120 steps. Per step: (16 x 192) x (192 x 384) GEMM
// via v_wmma_f32_16x16x32_bf16 (18 WMMA / wave). Hidden state lives in
// registers (fp32) + ping-pong bf16 LDS mirror for cross-wave A operands.
// One barrier per step; x tile for t+1 prefetched during step t.
// LDS = 156 KB -> two workgroups per 320 KB WGP.

typedef __attribute__((ext_vector_type(16))) __bf16 v16bf;
typedef __attribute__((ext_vector_type(8)))  float  v8f;

#define NB 4096
#define NT 120
#define NI 64
#define NH 128
#define NG 384  // 3*NH

__device__ __forceinline__ unsigned short f2bf(float f) {
  union { __bf16 b; unsigned short u; } cv;
  cv.b = (__bf16)f;  // RNE convert
  return cv.u;
}

__device__ __forceinline__ float sigmoid_f(float x) {
  return 1.0f / (1.0f + __expf(-x));
}
__device__ __forceinline__ float tanh_f(float x) {
  return 1.0f - 2.0f / (__expf(2.0f * x) + 1.0f);
}

union V16U { v16bf v; unsigned u[8]; };

// 16xK=32 bf16 A-operand from LDS (row-major, leading dim ldk elems).
// Contiguous K order (matches load_b): lanes 0-15 = rows; VGPR j holds
// K=2j,2j+1 (+16 for lanes 16-31).
__device__ __forceinline__ v16bf load_a(const unsigned short* base, int ldk,
                                        int lane, int k0) {
  const int row = lane & 15;
  const int hi  = lane >> 4;
  const unsigned* p = (const unsigned*)(base + row * ldk);  // bf16 pairs
  const int kb = (k0 >> 1) + hi * 8;
  V16U r;
#pragma unroll
  for (int j = 0; j < 4; ++j) {
    r.u[j]     = p[kb + j];
    r.u[4 + j] = p[kb + 4 + j];
  }
  return r.v;
}

// K=32 x N=16 bf16 B-operand from row-major weights in LDS: B[k][n] = W[gc+n][k].
// Lane = n + 16*hk, elements = K ascending -> one 32-byte contiguous read.
__device__ __forceinline__ v16bf load_b(const unsigned short* w, int ldk,
                                        int lane, int gc, int k0) {
  const int n  = lane & 15;
  const int hk = lane >> 4;
  return *(const v16bf*)(w + (gc + n) * ldk + k0 + hk * 16);
}

__device__ __forceinline__ v8f wmma_acc(v16bf a, v16bf b, v8f c) {
  return __builtin_amdgcn_wmma_f32_16x16x32_bf16(false, a, false, b, (short)0,
                                                 c, false, false);
}

__global__ __launch_bounds__(256) void gru_fused_kernel(
    const float* __restrict__ x,     // [B, T, I]
    const float* __restrict__ W_ih,  // [3H, I]
    const float* __restrict__ W_hh,  // [3H, H]
    const float* __restrict__ b_ih,  // [3H]
    const float* __restrict__ b_hh,  // [3H]
    float* __restrict__ out) {       // [B*T*H] outputs ++ [B*H] h_T
  __shared__ __align__(32) unsigned short s_wih[NG * NI];      // 48 KB
  __shared__ __align__(32) unsigned short s_whh[NG * NH];      // 96 KB
  __shared__ __align__(32) unsigned short s_x[2][16 * NI];     //  4 KB
  __shared__ __align__(32) unsigned short s_hb[2][16 * NH];    //  8 KB

  const int tid  = threadIdx.x;   // 8 waves of 32
  const int lane = tid & 31;
  const int wave = tid >> 5;      // owns hidden cols [16*wave, 16*wave+16)
  const int b0   = blockIdx.x * 16;

  // Stage weights (fp32 -> bf16) once; zero initial hidden mirror.
  for (int i = tid; i < NG * NI; i += 256) s_wih[i] = f2bf(W_ih[i]);
  for (int i = tid; i < NG * NH; i += 256) s_whh[i] = f2bf(W_hh[i]);
  for (int i = tid; i < 16 * NH; i += 256) s_hb[0][i] = 0;

  // Cooperative x-staging coordinates: 4 consecutive elems per thread.
  const int sidx = tid * 4;
  const int srow = sidx >> 6;   // batch row within tile
  const int scol = sidx & 63;   // input feature
  {  // stage x[:, 0, :]
    const float* xp = x + ((size_t)(b0 + srow) * NT + 0) * NI + scol;
    unsigned short* sp = &s_x[0][srow * NI + scol];
    sp[0] = f2bf(xp[0]); sp[1] = f2bf(xp[1]);
    sp[2] = f2bf(xp[2]); sp[3] = f2bf(xp[3]);
  }

  // Per-lane gate columns and biases (fixed across the T loop).
  const int col = lane & 15;
  const int hi  = lane >> 4;
  const int gr  = wave * 16;
  const int gz  = NH + wave * 16;
  const int gn  = 2 * NH + wave * 16;
  const float bias_r  = b_ih[gr + col] + b_hh[gr + col];
  const float bias_z  = b_ih[gz + col] + b_hh[gz + col];
  const float bias_xn = b_ih[gn + col];
  const float bias_hn = b_hh[gn + col];
  const int hcol = wave * 16 + col;

  v8f h_reg = {};  // persistent fp32 hidden tile (this wave's 16 columns)

  __syncthreads();

  for (int t = 0; t < NT; ++t) {
    const int p = t & 1;

    // Prefetch x tile for step t+1 (clamped re-read on the last step).
    const int tn = (t + 1 < NT) ? (t + 1) : (NT - 1);
    const float* xp = x + ((size_t)(b0 + srow) * NT + tn) * NI + scol;
    const float xf0 = xp[0], xf1 = xp[1], xf2 = xp[2], xf3 = xp[3];

    // A operands: x tile (K=64 -> 2), h tile (K=128 -> 4).
    const v16bf xA0 = load_a(s_x[p], NI, lane, 0);
    const v16bf xA1 = load_a(s_x[p], NI, lane, 32);
    const v16bf hA0 = load_a(s_hb[p], NH, lane, 0);
    const v16bf hA1 = load_a(s_hb[p], NH, lane, 32);
    const v16bf hA2 = load_a(s_hb[p], NH, lane, 64);
    const v16bf hA3 = load_a(s_hb[p], NH, lane, 96);

    v8f ar = {}, az = {}, axn = {}, ahn = {};

    // r gate: x@Wih_r + h@Whh_r fused into one accumulator.
    ar = wmma_acc(xA0, load_b(s_wih, NI, lane, gr, 0), ar);
    ar = wmma_acc(xA1, load_b(s_wih, NI, lane, gr, 32), ar);
    ar = wmma_acc(hA0, load_b(s_whh, NH, lane, gr, 0), ar);
    ar = wmma_acc(hA1, load_b(s_whh, NH, lane, gr, 32), ar);
    ar = wmma_acc(hA2, load_b(s_whh, NH, lane, gr, 64), ar);
    ar = wmma_acc(hA3, load_b(s_whh, NH, lane, gr, 96), ar);

    // z gate.
    az = wmma_acc(xA0, load_b(s_wih, NI, lane, gz, 0), az);
    az = wmma_acc(xA1, load_b(s_wih, NI, lane, gz, 32), az);
    az = wmma_acc(hA0, load_b(s_whh, NH, lane, gz, 0), az);
    az = wmma_acc(hA1, load_b(s_whh, NH, lane, gz, 32), az);
    az = wmma_acc(hA2, load_b(s_whh, NH, lane, gz, 64), az);
    az = wmma_acc(hA3, load_b(s_whh, NH, lane, gz, 96), az);

    // n gate: xn and hn stay separate (n = tanh(xn + r*hn)).
    axn = wmma_acc(xA0, load_b(s_wih, NI, lane, gn, 0), axn);
    axn = wmma_acc(xA1, load_b(s_wih, NI, lane, gn, 32), axn);
    ahn = wmma_acc(hA0, load_b(s_whh, NH, lane, gn, 0), ahn);
    ahn = wmma_acc(hA1, load_b(s_whh, NH, lane, gn, 32), ahn);
    ahn = wmma_acc(hA2, load_b(s_whh, NH, lane, gn, 64), ahn);
    ahn = wmma_acc(hA3, load_b(s_whh, NH, lane, gn, 96), ahn);

    // Dump prefetched x tile into the alternate buffer (read at t+1).
    {
      unsigned short* sp = &s_x[p ^ 1][srow * NI + scol];
      sp[0] = f2bf(xf0); sp[1] = f2bf(xf1);
      sp[2] = f2bf(xf2); sp[3] = f2bf(xf3);
    }

    // Gates + h update. D layout: VGPR v -> row v + 8*hi, col = lane&15.
#pragma unroll
    for (int v = 0; v < 8; ++v) {
      const int row = v + 8 * hi;
      const float rg = sigmoid_f(ar[v] + bias_r);
      const float zg = sigmoid_f(az[v] + bias_z);
      const float ng = tanh_f(axn[v] + bias_xn + rg * (ahn[v] + bias_hn));
      const float hnew = (1.0f - zg) * ng + zg * h_reg[v];
      h_reg[v] = hnew;
      s_hb[p ^ 1][row * NH + hcol] = f2bf(hnew);
      out[((size_t)(b0 + row) * NT + t) * NH + hcol] = hnew;
    }

    __syncthreads();  // publish s_hb[p^1] / s_x[p^1]; close WAR on buffer p
  }

  // Final hidden state h_T straight from registers.
  float* hT = out + (size_t)NB * NT * NH;
#pragma unroll
  for (int v = 0; v < 8; ++v) {
    hT[(size_t)(b0 + v + 8 * hi) * NH + hcol] = h_reg[v];
  }
}

extern "C" void kernel_launch(void* const* d_in, const int* in_sizes, int n_in,
                              void* d_out, int out_size, void* d_ws,
                              size_t ws_size, hipStream_t stream) {
  (void)in_sizes; (void)n_in; (void)d_ws; (void)ws_size; (void)out_size;
  const float* x    = (const float*)d_in[0];
  const float* W_ih = (const float*)d_in[1];
  const float* W_hh = (const float*)d_in[2];
  const float* b_ih = (const float*)d_in[3];
  const float* b_hh = (const float*)d_in[4];
  float* out = (float*)d_out;

  dim3 grid(NB / 16);   // 256 blocks of 16 batch rows
  dim3 block(256);      // 8 waves (wave32)
  gru_fused_kernel<<<grid, block, 0, stream>>>(x, W_ih, W_hh, b_ih, b_hh, out);
}
